// GATModel_61701500174761
// MI455X (gfx1250) — compile-verified, matrix operand order checked by reading, not run on hardware
//
#include <hip/hip_runtime.h>
#include <hip/hip_bf16.h>
#include <hip/hip_fp16.h>

// ---------------------------------------------------------------------------
// GAT (3 layers, 4/4/1 heads) + LN + ELU + MLP classifier for MI455X (gfx1250)
// - GEMMs: v_wmma_f32_16x16x32_f16, block = 8 waves covering 128x64 C tile,
//   B^T slice staged in LDS via the Tensor Data Mover (tensor_load_to_lds with
//   hardware row padding to avoid LDS bank conflicts), A fragment reused
//   across 4 WMMAs per K-step.
// - Edge softmax/aggregation: f32 global atomics; the 51 MB accumulator and
//   51 MB feature matrix both fit in the 192 MB L2 -> L2-resident scatter.
// ---------------------------------------------------------------------------

typedef __attribute__((ext_vector_type(16))) _Float16 v16h;
typedef __attribute__((ext_vector_type(8)))  _Float16 v8h;
typedef __attribute__((ext_vector_type(8)))  float    v8f;
typedef __attribute__((ext_vector_type(4)))  unsigned int v4u;
typedef __attribute__((ext_vector_type(4)))  int      v4i;
typedef __attribute__((ext_vector_type(8)))  int      v8i;

#define LEAKY_SLOPE 0.2f
#define LN_EPS_F    1e-5f
#define BPAD        8      // halfs of LDS row padding (16 B -> conflict-free)
#define MAXK        256

#define HAVE_TDM __has_builtin(__builtin_amdgcn_tensor_load_to_lds)

__device__ __forceinline__ float leakyf(float x) { return x > 0.f ? x : LEAKY_SLOPE * x; }
__device__ __forceinline__ float eluf(float x)   { return x > 0.f ? x : (__expf(x) - 1.f); }

// float atomic-max via monotone int/uint reinterpretation (init must be -inf)
__device__ __forceinline__ void atomicMaxF(float* addr, float v) {
  if (v >= 0.f) atomicMax((int*)addr, __float_as_int(v));
  else          atomicMin((unsigned int*)addr, __float_as_uint(v));
}

// ---------------------------------------------------------------------------
// generic fills / casts
// ---------------------------------------------------------------------------
__global__ void fill_f32(float* __restrict__ p, float v, int n) {
  int i = blockIdx.x * blockDim.x + threadIdx.x;
  if (i < n) p[i] = v;
}

__global__ void cast_f32_f16(const float* __restrict__ in, _Float16* __restrict__ out, int n) {
  int i = blockIdx.x * blockDim.x + threadIdx.x;
  if (i < n) out[i] = (_Float16)in[i];
}

// Wt[n*K + k] = W[k*Nc + n]
__global__ void cast_transpose_f32_f16(const float* __restrict__ W, _Float16* __restrict__ Wt,
                                       int K, int Nc) {
  int i = blockIdx.x * blockDim.x + threadIdx.x;
  if (i >= K * Nc) return;
  int k = i / Nc, n = i - k * Nc;
  Wt[(size_t)n * K + k] = (_Float16)W[i];
}

// ---------------------------------------------------------------------------
// WMMA GEMM: C[M,Nc] = A[M,K] * B[K,Nc]; A row-major f16, Bt = B^T row-major.
// Block = 256 threads (8 waves) -> 128 rows x 64 cols of C.
//   grid.x = ceil(M/128), grid.y = Nc/64.
// Bt slice (64 rows x K halfs) staged in LDS once per block via TDM with
// hardware padding (row stride K+BPAD halfs). Each wave: 1 M-tile x 4 N-tiles,
// A fragment loaded once per K-step and reused across 4 WMMAs.
// ---------------------------------------------------------------------------
__global__ void wmma_gemm_f16f32(const _Float16* __restrict__ A,
                                 const _Float16* __restrict__ Bt,
                                 float* __restrict__ C,
                                 int M, int Nc, int K) {
  __shared__ __align__(16) _Float16 sB[64 * (MAXK + BPAD)];
  const int mg   = blockIdx.x;
  const int ng   = blockIdx.y;
  const int wv   = threadIdx.x >> 5;
  const int lane = threadIdx.x & 31;
  const int l15  = lane & 15;
  const int hi   = lane >> 4;
  const int rowStride = K + BPAD;

  const _Float16* gB = Bt + (size_t)ng * 64 * K;   // 64 contiguous Bt rows

#if HAVE_TDM
  if (wv == 0) {
    // ---- D# group 0: count=1 | lds_addr | global_addr | type=2 ----
    unsigned long long q0lo = 0x1ull |
        ((unsigned long long)(unsigned)(size_t)&sB[0] << 32);
    unsigned long long q0hi =
        ((unsigned long long)(size_t)gB & 0x01FFFFFFFFFFFFFFull) | (2ull << 62);
    v4u g0;
    g0[0] = (unsigned)q0lo; g0[1] = (unsigned)(q0lo >> 32);
    g0[2] = (unsigned)q0hi; g0[3] = (unsigned)(q0hi >> 32);
    // ---- D# group 1: 2-D tile K x 64, data_size=2B, LDS row padding ----
    const unsigned padI = (K == 256) ? 6u : 5u;  // pad every K/2 dwords (one row)
    const unsigned padA = 3u;                    // 4 dwords = 16 B = BPAD halfs
    unsigned td0 = (unsigned)K, td1 = 64u, tl0 = (unsigned)K, tl1 = 64u;
    unsigned long long s0 = (unsigned long long)K;   // tensor_dim0_stride
    v8i g1;
    g1[0] = (int)((1u << 16) | (1u << 20) | (padI << 22) | (padA << 25));
    g1[1] = (int)((td0 & 0xFFFFu) << 16);                         // [63:48]=td0.lo
    g1[2] = (int)(((td0 >> 16) & 0xFFFFu) | ((td1 & 0xFFFFu) << 16));
    g1[3] = (int)(((td1 >> 16) & 0xFFFFu) | (tl0 << 16));         // tile_dim0
    g1[4] = (int)(tl1 & 0xFFFFu);                                  // tile_dim1
    g1[5] = (int)(unsigned)(s0 & 0xFFFFFFFFull);
    g1[6] = (int)(unsigned)((s0 >> 32) & 0xFFFFull);
    g1[7] = 0;
    v4i gz = {0, 0, 0, 0};
#if __clang_major__ >= 23
    v8i gz8 = {0, 0, 0, 0, 0, 0, 0, 0};
    __builtin_amdgcn_tensor_load_to_lds(g0, g1, gz, gz, gz8, 0);
#else
    __builtin_amdgcn_tensor_load_to_lds(g0, g1, gz, gz, 0);
#endif
    __builtin_amdgcn_s_wait_tensorcnt(0);
  }
#else
  // fallback: cooperative copy with manual row padding
  for (int idx = threadIdx.x; idx < 64 * (K >> 3); idx += 256) {
    int r  = idx / (K >> 3);
    int kk = (idx - r * (K >> 3)) * 8;
    *(v8h*)&sB[r * rowStride + kk] = *(const v8h*)(gB + (size_t)r * K + kk);
  }
#endif
  __syncthreads();

  const int tile_m = mg * 8 + wv;
  if (tile_m * 16 < M) {                 // wave-uniform guard: EXEC all-ones
    const _Float16* arow = A + (size_t)(tile_m * 16 + l15) * K;
    v8f acc[4] = {};
    for (int k0 = 0; k0 < K; k0 += 32) {
      if (k0 + 32 < K) __builtin_prefetch(arow + k0 + 32, 0, 3);
      v8h a0 = *(const v8h*)(arow + k0 + hi * 8);
      v8h a1 = *(const v8h*)(arow + k0 + 16 + hi * 8);
      v16h a;
#pragma unroll
      for (int i = 0; i < 8; ++i) { a[i] = a0[i]; a[i + 8] = a1[i]; }
#pragma unroll
      for (int t = 0; t < 4; ++t) {
        const _Float16* sp = &sB[(t * 16 + l15) * rowStride + k0 + hi * 8];
        v8h b0 = *(const v8h*)sp;
        v8h b1 = *(const v8h*)(sp + 16);
        v16h b;
#pragma unroll
        for (int i = 0; i < 8; ++i) { b[i] = b0[i]; b[i + 8] = b1[i]; }
        acc[t] = __builtin_amdgcn_wmma_f32_16x16x32_f16(false, a, false, b,
                                                        (short)0, acc[t], false, false);
      }
    }
#pragma unroll
    for (int t = 0; t < 4; ++t) {
      float* crow = C + (size_t)(tile_m * 16 + hi * 8) * Nc + ng * 64 + t * 16 + l15;
#pragma unroll
      for (int r = 0; r < 8; ++r) crow[(size_t)r * Nc] = acc[t][r];
    }
  }
}

// ---------------------------------------------------------------------------
// attention coefficients: a_src[n,h] = <h[n,h,:], att_src[h,:]>, same for dst
// ---------------------------------------------------------------------------
__global__ void attn_coef(const float* __restrict__ hmat,
                          const float* __restrict__ att_src,
                          const float* __restrict__ att_dst,
                          float* __restrict__ a_src, float* __restrict__ a_dst,
                          int N, int H, int C) {
  int idx = blockIdx.x * blockDim.x + threadIdx.x;
  if (idx >= N * H) return;
  int n = idx / H, h = idx - n * H;
  const float* hp = hmat + (size_t)n * H * C + (size_t)h * C;
  const float* ws = att_src + (size_t)h * C;
  const float* wd = att_dst + (size_t)h * C;
  float ss = 0.f, sd = 0.f;
  for (int c = 0; c < C; ++c) { float v = hp[c]; ss += v * ws[c]; sd += v * wd[c]; }
  a_src[idx] = ss;
  a_dst[idx] = sd;
}

// ---------------------------------------------------------------------------
// edge passes (Et = E real edges + N self-loops appended)
// ---------------------------------------------------------------------------
__global__ void edge_seg_max(const int* __restrict__ ei, int E, int N,
                             const float* __restrict__ a_src,
                             const float* __restrict__ a_dst,
                             float* __restrict__ seg_m, int H) {
  int e = blockIdx.x * blockDim.x + threadIdx.x;
  if (e >= E + N) return;
  int s = (e < E) ? ei[e]     : (e - E);
  int d = (e < E) ? ei[E + e] : (e - E);
  for (int h = 0; h < H; ++h) {
    float l = leakyf(a_src[s * H + h] + a_dst[d * H + h]);
    atomicMaxF(&seg_m[d * H + h], l);
  }
}

__global__ void edge_seg_expsum(const int* __restrict__ ei, int E, int N,
                                const float* __restrict__ a_src,
                                const float* __restrict__ a_dst,
                                const float* __restrict__ seg_m,
                                float* __restrict__ seg_d, int H) {
  int e = blockIdx.x * blockDim.x + threadIdx.x;
  if (e >= E + N) return;
  int s = (e < E) ? ei[e]     : (e - E);
  int d = (e < E) ? ei[E + e] : (e - E);
  for (int h = 0; h < H; ++h) {
    float m = seg_m[d * H + h];
    if (!(m > -1e30f)) m = 0.f;          // isfinite guard from reference
    float l = leakyf(a_src[s * H + h] + a_dst[d * H + h]);
    atomicAdd(&seg_d[d * H + h], __expf(l - m));
  }
}

// one wave per edge; lanes 0..H-1 compute alpha, broadcast via shfl, lanes
// sweep the H*C channels with L2-resident f32 atomic adds
__global__ void edge_aggregate(const int* __restrict__ ei, int E, int N,
                               const float* __restrict__ hmat,
                               const float* __restrict__ a_src,
                               const float* __restrict__ a_dst,
                               const float* __restrict__ seg_m,
                               const float* __restrict__ seg_d,
                               float* __restrict__ out, int H, int C) {
  int wave = (blockIdx.x * blockDim.x + threadIdx.x) >> 5;
  int lane = threadIdx.x & 31;
  if (wave >= E + N) return;
  int s = (wave < E) ? ei[wave]     : (wave - E);
  int d = (wave < E) ? ei[E + wave] : (wave - E);
  const int D = H * C;
  const float* hs = hmat + (size_t)s * D;
  float*       od = out  + (size_t)d * D;

  float aval = 0.f;
  if (lane < H) {
    float m = seg_m[d * H + lane];
    if (!(m > -1e30f)) m = 0.f;
    float l = leakyf(a_src[s * H + lane] + a_dst[d * H + lane]);
    aval = __expf(l - m) / (seg_d[d * H + lane] + 1e-16f);
  }
  for (int j = lane; j < D; j += 32) {
    float alpha = __shfl(aval, j / C, 32);
    atomicAdd(&od[j], alpha * hs[j]);
  }
}

// ---------------------------------------------------------------------------
// bias + LayerNorm + ELU, writes f16 operand for the next GEMM. blockDim == D.
// ---------------------------------------------------------------------------
__global__ void post_ln_elu(const float* __restrict__ agg,
                            const float* __restrict__ bias,
                            const float* __restrict__ gamma,
                            const float* __restrict__ beta,
                            _Float16* __restrict__ out16, int D) {
  int n = blockIdx.x;
  int t = threadIdx.x;
  float v = agg[(size_t)n * D + t] + bias[t];
  __shared__ float s1[256];
  __shared__ float s2[256];
  s1[t] = v; s2[t] = v * v;
  __syncthreads();
  for (int stride = 128; stride > 0; stride >>= 1) {
    if (t < stride) { s1[t] += s1[t + stride]; s2[t] += s2[t + stride]; }
    __syncthreads();
  }
  float mu  = s1[0] / (float)D;
  float var = s2[0] / (float)D - mu * mu;
  float y = (v - mu) * rsqrtf(var + LN_EPS_F) * gamma[t] + beta[t];
  out16[(size_t)n * D + t] = (_Float16)eluf(y);
}

// ---------------------------------------------------------------------------
// classifier tail: ELU(hidden + cb1) @ cW2 + cb2  (Nc=2: plain VALU kernel)
// ---------------------------------------------------------------------------
__global__ void final_mlp(const float* __restrict__ c1,
                          const float* __restrict__ cb1,
                          const float* __restrict__ cW2,
                          const float* __restrict__ cb2,
                          float* __restrict__ out, int N) {
  int n = blockIdx.x * blockDim.x + threadIdx.x;
  if (n >= N) return;
  float acc0 = cb2[0], acc1 = cb2[1];
  const float* row = c1 + (size_t)n * 128;
  for (int k = 0; k < 128; ++k) {
    float v = eluf(row[k] + cb1[k]);
    acc0 += v * cW2[k * 2 + 0];
    acc1 += v * cW2[k * 2 + 1];
  }
  out[n * 2 + 0] = acc0;
  out[n * 2 + 1] = acc1;
}

// ---------------------------------------------------------------------------
extern "C" void kernel_launch(void* const* d_in, const int* in_sizes, int n_in,
                              void* d_out, int out_size, void* d_ws, size_t ws_size,
                              hipStream_t stream) {
  const float* x    = (const float*)d_in[0];
  const int*   ei   = (const int*)d_in[1];
  const float* W0   = (const float*)d_in[2];
  const float* as0  = (const float*)d_in[3];
  const float* ad0  = (const float*)d_in[4];
  const float* b0   = (const float*)d_in[5];
  const float* W1   = (const float*)d_in[6];
  const float* as1  = (const float*)d_in[7];
  const float* ad1  = (const float*)d_in[8];
  const float* b1   = (const float*)d_in[9];
  const float* W2   = (const float*)d_in[10];
  const float* as2  = (const float*)d_in[11];
  const float* ad2  = (const float*)d_in[12];
  const float* b2   = (const float*)d_in[13];
  const float* lng0 = (const float*)d_in[14];
  const float* lnb0 = (const float*)d_in[15];
  const float* lng1 = (const float*)d_in[16];
  const float* lnb1 = (const float*)d_in[17];
  const float* lng2 = (const float*)d_in[18];
  const float* lnb2 = (const float*)d_in[19];
  const float* cW1  = (const float*)d_in[20];
  const float* cb1  = (const float*)d_in[21];
  const float* cW2  = (const float*)d_in[22];
  const float* cb2  = (const float*)d_in[23];
  float* out = (float*)d_out;

  const int N  = in_sizes[0] / 128;   // 50000
  const int E  = in_sizes[1] / 2;     // 800000
  const int Et = E + N;               // with self-loops
  const int HID = 256;

  // ---- workspace carve-up (~130 MB) ----
  char* ws = (char*)d_ws;
  size_t off = 0;
  auto take = [&](size_t bytes) -> char* {
    char* p = ws + off;
    off = (off + bytes + 255) & ~(size_t)255;
    return p;
  };
  _Float16* hA16  = (_Float16*)take((size_t)N * HID * sizeof(_Float16));
  _Float16* Wt16  = (_Float16*)take((size_t)HID * HID * sizeof(_Float16));
  float*    hH    = (float*)take((size_t)N * HID * sizeof(float));
  float*    hAgg  = (float*)take((size_t)N * HID * sizeof(float));
  float*    aSrc  = (float*)take((size_t)N * 4 * sizeof(float));
  float*    aDst  = (float*)take((size_t)N * 4 * sizeof(float));
  float*    segM  = (float*)take((size_t)N * 4 * sizeof(float));
  float*    segD  = (float*)take((size_t)N * 4 * sizeof(float));

  const int TPB = 256;
  auto blocks = [](int n, int tpb) { return (n + tpb - 1) / tpb; };
  const int mGroups = (N / 16 + 7) / 8;   // 8 M-tiles per block

  auto run_gat_layer = [&](int K, const float* W, const float* aws, const float* awd,
                           const float* bias, const float* g, const float* b,
                           int H, int C) {
    const int D = H * C;  // 256 for all layers
    cast_transpose_f32_f16<<<blocks(K * D, TPB), TPB, 0, stream>>>(W, Wt16, K, D);
    dim3 gg(mGroups, D / 64);
    wmma_gemm_f16f32<<<gg, TPB, 0, stream>>>(hA16, Wt16, hH, N, D, K);
    attn_coef<<<blocks(N * H, TPB), TPB, 0, stream>>>(hH, aws, awd, aSrc, aDst, N, H, C);
    fill_f32<<<blocks(N * H, TPB), TPB, 0, stream>>>(segM, -INFINITY, N * H);
    fill_f32<<<blocks(N * H, TPB), TPB, 0, stream>>>(segD, 0.f, N * H);
    fill_f32<<<blocks(N * D, TPB), TPB, 0, stream>>>(hAgg, 0.f, N * D);
    edge_seg_max<<<blocks(Et, TPB), TPB, 0, stream>>>(ei, E, N, aSrc, aDst, segM, H);
    edge_seg_expsum<<<blocks(Et, TPB), TPB, 0, stream>>>(ei, E, N, aSrc, aDst, segM, segD, H);
    edge_aggregate<<<blocks(Et * 32, TPB), TPB, 0, stream>>>(ei, E, N, hH, aSrc, aDst,
                                                             segM, segD, hAgg, H, C);
    post_ln_elu<<<N, 256, 0, stream>>>(hAgg, bias, g, b, hA16, D);
  };

  // layer 0 input: cast x [N,128] to f16
  cast_f32_f16<<<blocks(N * 128, TPB), TPB, 0, stream>>>(x, hA16, N * 128);
  run_gat_layer(128, W0, as0, ad0, b0, lng0, lnb0, 4, 64);
  run_gat_layer(256, W1, as1, ad1, b1, lng1, lnb1, 4, 64);
  run_gat_layer(256, W2, as2, ad2, b2, lng2, lnb2, 1, 256);

  // classifier: hidden = hA16 @ cW1 (WMMA, Nc=128), then ELU + 128->2 projection
  cast_transpose_f32_f16<<<blocks(256 * 128, TPB), TPB, 0, stream>>>(cW1, Wt16, 256, 128);
  dim3 gc(mGroups, 128 / 64);
  wmma_gemm_f16f32<<<gc, TPB, 0, stream>>>(hA16, Wt16, hH, N, 128, 256);
  final_mlp<<<blocks(N, TPB), TPB, 0, stream>>>(hH, cb1, cW2, cb2, out, N);
}
